// SelfAttention_35880156790799
// MI455X (gfx1250) — matrix operand order repaired and checked by reading, not compiled, not run on hardware
//
#include <hip/hip_runtime.h>
#include <math.h>

#define B 4
#define H 8
#define T 4096
#define D 64
#define NC 16
#define WSZ 256
#define EPS 1e-5f

// padded LDS strides (in halfs) to avoid bank conflicts on b128 lane-strided reads
#define KSTRIDE 72    // 144 B rows for gathered K
#define VSTRIDE 264   // 528 B rows for transposed V
#define PSTRIDE 264   // 528 B rows for probability strips

typedef _Float16 half_t;
typedef __attribute__((ext_vector_type(8)))  _Float16 v8h;
typedef __attribute__((ext_vector_type(16))) _Float16 v16h;
typedef __attribute__((ext_vector_type(8)))  float    v8f;

__device__ __forceinline__ v8f wmma_f16(v16h a, v16h b, v8f c) {
  // D = A(16x32 f16) * B(32x16 f16) + C(16x16 f32)
  return __builtin_amdgcn_wmma_f32_16x16x32_f16(
      /*neg_a=*/false, a, /*neg_b=*/false, b,
      /*c_mod=*/(short)0, c, /*reuse_a=*/false, /*reuse_b=*/false);
}

__device__ __forceinline__ v16h cat8(v8h lo, v8h hi) {
  return __builtin_shufflevector(lo, hi, 0, 1, 2, 3, 4, 5, 6, 7,
                                         8, 9, 10, 11, 12, 13, 14, 15);
}

// ---------------- 1. x = l2norm(k) ----------------
__global__ void l2norm_kernel(const float* __restrict__ k, float* __restrict__ x) {
  int t = blockIdx.x * blockDim.x + threadIdx.x;
  if (t >= B * H * T) return;
  const float* kr = k + (size_t)t * D;
  float s = 0.f;
  for (int d = 0; d < D; ++d) { float v = kr[d]; s += v * v; }
  float inv = 1.0f / fmaxf(sqrtf(s), 1e-12f);
  float* xr = x + (size_t)t * D;
  for (int d = 0; d < D; ++d) xr[d] = kr[d] * inv;
}

// ---------------- 2. dists = x @ means[h]^T via WMMA ----------------
// One wave handles one 16-row tile, N = 16 clusters (single tile), K = 64 (2 steps)
__global__ __launch_bounds__(256) void dists_kernel(const float* __restrict__ x,
                                                    const float* __restrict__ means,
                                                    float* __restrict__ dists) {
  int wave = threadIdx.x >> 5, lane = threadIdx.x & 31;
  int gw   = blockIdx.x * 8 + wave;       // global wave id
  int bh   = gw >> 8;                     // T/16 = 256 tiles per (b,h)
  int tile = gw & 255;
  int h    = bh & (H - 1);
  int m = lane & 15, hs = lane >> 4;
  const float* xr = x + ((size_t)bh * T + tile * 16 + m) * D;   // A row (M = m)
  const float* mr = means + ((size_t)h * NC + m) * D;           // B col (N = lane&15)
  v8f c = {};
  for (int kb = 0; kb < 64; kb += 32) {
    v16h a, bm;
#pragma unroll
    for (int t = 0; t < 8; ++t) {
      a[t]     = (half_t)xr[kb + 16 * hs + t];         // K = kb + 16*hs + t
      a[8 + t] = (half_t)xr[kb + 16 * hs + 8 + t];     // K = kb + 16*hs + 8 + t
    }
#pragma unroll
    for (int t = 0; t < 16; ++t)
      bm[t] = (half_t)mr[kb + hs * 16 + t];            // B[K=kb+hs*16+t][n]
    c = wmma_f16(a, bm, c);
  }
  float* dr = dists + ((size_t)bh * T + tile * 16) * NC;
#pragma unroll
  for (int r = 0; r < 8; ++r)
    dr[(r + 8 * hs) * NC + m] = c[r];                  // C row = r + 8*hs, col = lane&15
}

// ---------------- 3. per (b,h,c): top-256 tokens, indices sorted ascending ----------------
__global__ __launch_bounds__(256) void topk_kernel(const float* __restrict__ dists,
                                                   int* __restrict__ idx_out) {
  __shared__ float sval[T];
  __shared__ int   sidx[T];
  int blk = blockIdx.x;
  int c = blk & (NC - 1), bh = blk >> 4;
  const float* dp = dists + (size_t)bh * T * NC + c;
  for (int i = threadIdx.x; i < T; i += 256) { sval[i] = dp[(size_t)i * NC]; sidx[i] = i; }
  __syncthreads();
  // bitonic sort: descending by value, ties -> lower index first (matches top_k)
  for (int k = 2; k <= T; k <<= 1) {
    for (int j = k >> 1; j > 0; j >>= 1) {
      for (int i = threadIdx.x; i < T; i += 256) {
        int ixj = i ^ j;
        if (ixj > i) {
          float v0 = sval[i], v1 = sval[ixj];
          int   i0 = sidx[i], i1 = sidx[ixj];
          bool before = (v0 > v1) || (v0 == v1 && i0 < i1);
          bool up = ((i & k) == 0);
          if (up ? !before : before) {
            sval[i] = v1; sval[ixj] = v0; sidx[i] = i1; sidx[ixj] = i0;
          }
        }
      }
      __syncthreads();
    }
  }
  // sort top-256 indices ascending (in place over sidx[0..255])
  for (int k = 2; k <= WSZ; k <<= 1) {
    for (int j = k >> 1; j > 0; j >>= 1) {
      int i = threadIdx.x, ixj = i ^ j;
      if (ixj > i) {
        int i0 = sidx[i], i1 = sidx[ixj];
        bool asc = ((i & k) == 0);
        if (asc ? (i0 > i1) : (i0 < i1)) { sidx[i] = i1; sidx[ixj] = i0; }
      }
      __syncthreads();
    }
  }
  idx_out[(size_t)blk * WSZ + threadIdx.x] = sidx[threadIdx.x];
}

// ---------------- 4. windowed attention via WMMA ----------------
__device__ __forceinline__ float masked_score(float dot, int wi, int wj) {
  float s = dot * 0.125f;               // d^-0.5 = 1/8
  if (wj > wi)       s = -3.0e38f;      // causal (window positions)
  else if (wj == wi) s = -50000.0f;     // shared-qk self-token mask
  return s;
}

__device__ __forceinline__ v8f score_tile(v16h aq0, v16h aq1, const half_t* kf,
                                          int ct, int m, int hs) {
  const half_t* kj = kf + (ct * 16 + m) * KSTRIDE;   // key row j = ct*16 + n (n = lane&15)
  // B[K=d][n]: 8-half runs at 8*hs and 16+8*hs -> ds_load_b128 pairs
  v16h b0 = cat8(*(const v8h*)(kj + hs * 8),      *(const v8h*)(kj + 16 + hs * 8));
  v16h b1 = cat8(*(const v8h*)(kj + 32 + hs * 8), *(const v8h*)(kj + 48 + hs * 8));
  v8f c = {};
  c = wmma_f16(aq0, b0, c);
  c = wmma_f16(aq1, b1, c);
  return c;
}

__global__ __launch_bounds__(256) void attn_kernel(const float* __restrict__ q,
                                                   const float* __restrict__ xk,
                                                   const float* __restrict__ v,
                                                   const int* __restrict__ idx,
                                                   float* __restrict__ so) {
  __shared__ int    sidx[WSZ];                 //  1 KB
  __shared__ half_t kf[WSZ * KSTRIDE];         // 36 KB : gathered K rows (padded)
  __shared__ half_t vT[D * VSTRIDE];           // 33 KB : gathered V, transposed [d][j]
  __shared__ half_t pstrip[8 * 16 * PSTRIDE];  // 66 KB : per-wave 16x256 prob strips
  int win = blockIdx.x;
  int bh  = win >> 4;
  size_t base = (size_t)bh * T * D;
  int tid = threadIdx.x;
  sidx[tid] = idx[(size_t)win * WSZ + tid];
  __syncthreads();
  for (int e = tid; e < WSZ * D; e += 256) {
    int r = e >> 6, d = e & 63;
    size_t g = base + (size_t)sidx[r] * D + d;
    kf[r * KSTRIDE + d] = (half_t)xk[g];
    vT[d * VSTRIDE + r] = (half_t)v[g];
  }
  __syncthreads();
  int wave = tid >> 5, lane = tid & 31;
  int m = lane & 15, hs = lane >> 4;
  half_t* ps = pstrip + wave * (16 * PSTRIDE);
  for (int rb = wave; rb < 16; rb += 8) {     // each wave: 2 row-blocks of 16 queries
    const float* qp = q + base + (size_t)sidx[rb * 16 + m] * D;
    v16h aq0, aq1;
#pragma unroll
    for (int t = 0; t < 8; ++t) {
      aq0[t]     = (half_t)qp[8 * hs + t];
      aq0[8 + t] = (half_t)qp[16 + 8 * hs + t];
      aq1[t]     = (half_t)qp[32 + 8 * hs + t];
      aq1[8 + t] = (half_t)qp[48 + 8 * hs + t];
    }
    float rmax[8], rsum[8];
#pragma unroll
    for (int r = 0; r < 8; ++r) { rmax[r] = -3.0e38f; rsum[r] = 0.f; }
    // pass 1: row max
    for (int ct = 0; ct < 16; ++ct) {
      v8f c = score_tile(aq0, aq1, kf, ct, m, hs);
#pragma unroll
      for (int r = 0; r < 8; ++r)
        rmax[r] = fmaxf(rmax[r], masked_score(c[r], rb * 16 + r + 8 * hs, ct * 16 + m));
    }
#pragma unroll
    for (int off = 1; off < 16; off <<= 1)
#pragma unroll
      for (int r = 0; r < 8; ++r)
        rmax[r] = fmaxf(rmax[r], __shfl_xor(rmax[r], off, 32));
    // pass 2: recompute, exponentiate, stage probs (f16), accumulate row sums
    for (int ct = 0; ct < 16; ++ct) {
      v8f c = score_tile(aq0, aq1, kf, ct, m, hs);
#pragma unroll
      for (int r = 0; r < 8; ++r) {
        float s = masked_score(c[r], rb * 16 + r + 8 * hs, ct * 16 + m);
        float p = __expf(s - rmax[r]);
        rsum[r] += p;
        ps[(r + 8 * hs) * PSTRIDE + ct * 16 + m] = (half_t)p;
      }
    }
#pragma unroll
    for (int off = 1; off < 16; off <<= 1)
#pragma unroll
      for (int r = 0; r < 8; ++r)
        rsum[r] += __shfl_xor(rsum[r], off, 32);
    // pass 3: out = (P @ V) / rowsum
    float* sop = so + ((size_t)win * WSZ + rb * 16) * D;
    for (int dt = 0; dt < 4; ++dt) {
      v8f c = {};
      for (int kt = 0; kt < 8; ++kt) {
        int kb = kt * 32;
        const half_t* pr = ps + m * PSTRIDE + kb;                 // A row m, K block
        v16h ap = cat8(*(const v8h*)(pr + 16 * hs),
                       *(const v8h*)(pr + 16 * hs + 8));
        const half_t* vr = vT + (dt * 16 + m) * VSTRIDE + kb + hs * 16;  // B[K=j][n=d']
        v16h bv = cat8(*(const v8h*)vr, *(const v8h*)(vr + 8));
        c = wmma_f16(ap, bv, c);
      }
#pragma unroll
      for (int r = 0; r < 8; ++r)
        sop[(r + 8 * hs) * D + dt * 16 + m] = c[r] / rsum[r];
    }
  }
}

// ---------------- 5. scatter-mean (gather-side, deterministic, atomic-free) ----------------
__global__ void scatter_kernel(const int* __restrict__ idx, const float* __restrict__ so,
                               float* __restrict__ out) {
  size_t g = (size_t)blockIdx.x * blockDim.x + threadIdx.x;
  if (g >= (size_t)B * H * T * D) return;
  int d  = (int)(g & 63);
  int t  = (int)((g >> 6) & (T - 1));
  int bh = (int)(g >> 18);
  float num = 0.f, den = 0.f;
  for (int c = 0; c < NC; ++c) {
    const int* ip = idx + ((size_t)bh * NC + c) * WSZ;
    int lo = 0, hi = WSZ - 1, pos = -1;
    while (lo <= hi) {
      int mid = (lo + hi) >> 1;
      int vv = ip[mid];
      if (vv == t) { pos = mid; break; }
      else if (vv < t) lo = mid + 1;
      else hi = mid - 1;
    }
    if (pos >= 0) {
      num += so[(((size_t)bh * NC + c) * WSZ + pos) * D + d];
      den += 1.f;
    }
  }
  out[g] = num / (den + EPS);
}

// ---------------- 6. commitment loss (two-stage deterministic reduction) ----------------
__global__ __launch_bounds__(256) void aux_partial_kernel(const float* __restrict__ x,
                                                          const float* __restrict__ dists,
                                                          const float* __restrict__ means,
                                                          float* __restrict__ partial) {
  __shared__ float red[256];
  int tok = blockIdx.x * 256 + threadIdx.x;
  float s = 0.f;
  if (tok < B * H * T) {
    int bh = tok >> 12;
    int h  = bh & (H - 1);
    const float* dp = dists + (size_t)tok * NC;
    float best = dp[0]; int bc = 0;
    for (int c = 1; c < NC; ++c) { float vv = dp[c]; if (vv > best) { best = vv; bc = c; } }
    const float* xr = x + (size_t)tok * D;
    const float* mr = means + ((size_t)h * NC + bc) * D;
    for (int d = 0; d < D; ++d) { float df = xr[d] - mr[d]; s += df * df; }
  }
  red[threadIdx.x] = s;
  __syncthreads();
  for (int st = 128; st > 0; st >>= 1) {
    if (threadIdx.x < st) red[threadIdx.x] += red[threadIdx.x + st];
    __syncthreads();
  }
  if (threadIdx.x == 0) partial[blockIdx.x] = red[0];
}

__global__ __launch_bounds__(256) void aux_final_kernel(const float* __restrict__ partial,
                                                        float* __restrict__ out_loss) {
  __shared__ float red[256];
  red[threadIdx.x] = partial[threadIdx.x] + partial[threadIdx.x + 256];
  __syncthreads();
  for (int st = 128; st > 0; st >>= 1) {
    if (threadIdx.x < st) red[threadIdx.x] += red[threadIdx.x + st];
    __syncthreads();
  }
  if (threadIdx.x == 0)
    out_loss[0] = red[0] * (1e-4f / (float)((size_t)B * H * T * D));
}

extern "C" void kernel_launch(void* const* d_in, const int* in_sizes, int n_in,
                              void* d_out, int out_size, void* d_ws, size_t ws_size,
                              hipStream_t stream) {
  (void)in_sizes; (void)n_in; (void)out_size; (void)ws_size;
  const float* q     = (const float*)d_in[0];
  const float* k     = (const float*)d_in[1];
  const float* v     = (const float*)d_in[2];
  const float* means = (const float*)d_in[3];
  float* out = (float*)d_out;

  char* ws = (char*)d_ws;
  float* x      = (float*)ws; ws += (size_t)B * H * T * D * sizeof(float);       // 33.5 MB
  float* dists  = (float*)ws; ws += (size_t)B * H * T * NC * sizeof(float);      // 8.4 MB
  int*   idx    = (int*)  ws; ws += (size_t)B * H * NC * WSZ * sizeof(int);      // 0.5 MB
  float* so     = (float*)ws; ws += (size_t)B * H * NC * WSZ * D * sizeof(float);// 33.5 MB
  float* partial= (float*)ws;                                                    // 2 KB

  l2norm_kernel<<<(B * H * T + 255) / 256, 256, 0, stream>>>(k, x);
  dists_kernel<<<(B * H * (T / 16)) / 8, 256, 0, stream>>>(x, means, dists);
  topk_kernel<<<B * H * NC, 256, 0, stream>>>(dists, idx);
  attn_kernel<<<B * H * NC, 256, 0, stream>>>(q, x, v, idx, so);
  scatter_kernel<<<(int)(((size_t)B * H * T * D + 255) / 256), 256, 0, stream>>>(idx, so, out);
  aux_partial_kernel<<<(B * H * T) / 256, 256, 0, stream>>>(x, dists, means, partial);
  aux_final_kernel<<<1, 256, 0, stream>>>(partial, out + (size_t)B * H * T * D);
}
